// SparseBlock_Conv2d_BN_ReLU_28389733826726
// MI455X (gfx1250) — matrix operand order repaired and checked by reading, not compiled, not run on hardware
//
#include <hip/hip_runtime.h>
#include <hip/hip_bf16.h>

// ---------------- types for WMMA (CDNA5, wave32) ----------------
typedef __bf16 bf16_t;
typedef __attribute__((ext_vector_type(16))) bf16_t v16bf;
typedef __attribute__((ext_vector_type(8)))  float  v8f;

union FragBF {            // 16 bf16 = 32 bytes = 8 VGPRs
    v16bf v;
    uint4 u[2];
};

#define NBLK   800        // active blocks (M in reference)
#define BCNT   40
#define BSP    22         // patch side
#define BOUTP  20         // output tile side
#define CH     64         // CIN == COUT
#define KTOT   576        // 3*3*64
#define KSTEPS 18         // KTOT / 32
#define HW     800
#define EPSV   1e-3f

__device__ __forceinline__ unsigned int bf_round(float f) {
    unsigned int u = __float_as_uint(f);
    u += 0x7FFFu + ((u >> 16) & 1u);          // round-to-nearest-even
    return u >> 16;
}

// ---------------- phase 0: repack weights to bf16, [cout][k] ----------------
__global__ __launch_bounds__(256) void pack_weights(const float* __restrict__ w,
                                                    unsigned short* __restrict__ Wt) {
    int i = blockIdx.x * 256 + threadIdx.x;           // over 576*64
    if (i >= KTOT * CH) return;
    int co = i & (CH - 1);
    int k  = i >> 6;                                   // (ky*3+kx)*64 + ci
    Wt[co * KTOT + k] = (unsigned short)bf_round(w[k * CH + co]);
}

// ---------------- phase 1: per-block implicit-GEMM conv + stats ----------------
__global__ __launch_bounds__(256) void conv_wmma_kernel(const float* __restrict__ inp,
                                                        const int*   __restrict__ bidx,
                                                        const unsigned short* __restrict__ Wt,
                                                        const float* __restrict__ conv_b,
                                                        float* __restrict__ out,
                                                        float* __restrict__ stats) {
    __shared__ __align__(16) unsigned short patch[BSP * BSP * CH];  // bf16 tile, 60.5 KB
    __shared__ float statsLDS[2 * CH];

    const int tid  = threadIdx.x;
    const int bb   = blockIdx.x;
    const int bh   = bidx[bb * 3 + 1];
    const int bw   = bidx[bb * 3 + 2];

    if (tid < 2 * CH) statsLDS[tid] = 0.0f;

    // ---- stage 22x22x64 fp32 patch -> bf16 LDS (pairwise, coalesced) ----
    const int y0 = bh * BOUTP - 1;
    const int x0 = bw * BOUTP - 1;
    for (int e = tid; e < BSP * BSP * (CH / 2); e += 256) {
        int cp = e & 31;               // channel pair
        int xy = e >> 5;
        int x  = xy % BSP;
        int y  = xy / BSP;
        int gy = y0 + y, gx = x0 + x;
        float2 v = make_float2(0.0f, 0.0f);
        if (gy >= 0 && gy < HW && gx >= 0 && gx < HW)
            v = *(const float2*)(inp + ((size_t)gy * HW + gx) * CH + cp * 2);
        unsigned int pk = bf_round(v.x) | (bf_round(v.y) << 16);
        *(unsigned int*)(patch + (y * BSP + x) * CH + cp * 2) = pk;
    }
    __syncthreads();

    const int wave  = tid >> 5;
    const int lane  = tid & 31;
    const int mrow  = lane & 15;       // A-matrix row / C-matrix column index
    const int khalf = lane >> 4;       // 0: K 0-7,16-23; 1: K 8-15,24-31

    for (int mt = wave; mt < 25; mt += 8) {      // 25 M-tiles of 16 pixels
        v8f zero = {0.f, 0.f, 0.f, 0.f, 0.f, 0.f, 0.f, 0.f};
        v8f acc[4] = {zero, zero, zero, zero};

        const int p  = mt * 16 + mrow;           // this lane's pixel (A row)
        const int oy = p / BOUTP;
        const int ox = p % BOUTP;

        for (int ks = 0; ks < KSTEPS; ++ks) {
            const int tap = ks >> 1;             // 0..8, never split by 32-wide K step
            const int ci0 = (ks & 1) << 5;
            const int ky  = tap / 3;
            const int kx  = tap % 3;

            // A fragment: lanes 0-15 hold K = {0..7,16..23}, lanes 16-31 the +8 offsets
            const int abase = ((oy + ky) * BSP + (ox + kx)) * CH + ci0 + khalf * 8;
            FragBF a;
            a.u[0] = *(const uint4*)(patch + abase);        // ds_load_b128
            a.u[1] = *(const uint4*)(patch + abase + 16);   // ds_load_b128

            const int kb = tap * CH + ci0 + khalf * 8;      // B: same K order as A
#pragma unroll
            for (int nt = 0; nt < 4; ++nt) {
                const unsigned short* wp = Wt + (nt * 16 + mrow) * KTOT + kb;
                FragBF b;
                b.u[0] = *(const uint4*)(wp);               // global_load_b128
                b.u[1] = *(const uint4*)(wp + 16);
                acc[nt] = __builtin_amdgcn_wmma_f32_16x16x32_bf16(
                    false, a.v, false, b.v, (short)0, acc[nt], false, false);
            }
        }

        // ---- epilogue: bias, write q to canvas, local BN stats ----
        const int mhi = khalf * 8;               // C/D rows m = r + 8*(lane>>4)
#pragma unroll
        for (int nt = 0; nt < 4; ++nt) {
            const int co = nt * 16 + mrow;       // N = lane & 15
            const float bias = conv_b[co];
            float s = 0.0f, s2 = 0.0f;
#pragma unroll
            for (int r = 0; r < 8; ++r) {
                const int pp = mt * 16 + r + mhi;
                const int qy = pp / BOUTP;
                const int qx = pp % BOUTP;
                const float v = acc[nt][r] + bias;
                s  += v;
                s2 += v * v;
                out[((size_t)(bh * BOUTP + qy) * HW + (bw * BOUTP + qx)) * CH + co] = v;
            }
            atomicAdd(&statsLDS[co], s);         // ds_add_f32
            atomicAdd(&statsLDS[CH + co], s2);
        }
    }
    __syncthreads();
    if (tid < 2 * CH) atomicAdd(&stats[tid], statsLDS[tid]);
}

// ---------------- phase 2: fold stats into scale/shift ----------------
__global__ void bn_stats_kernel(const float* __restrict__ stats,
                                const float* __restrict__ gamma,
                                const float* __restrict__ beta,
                                float* __restrict__ scsh) {
    int c = threadIdx.x;                          // 64 threads
    const float n  = (float)NBLK * (float)(BOUTP * BOUTP);
    float mu  = stats[c] / n;
    float var = stats[CH + c] / n - mu * mu;
    float sc  = gamma[c] * rsqrtf(var + EPSV);
    scsh[c]       = sc;
    scsh[CH + c]  = beta[c] - mu * sc;
}

// ---------------- phase 3: in-place normalize + ReLU over active tiles ----------------
__global__ __launch_bounds__(256) void bn_relu_kernel(const int* __restrict__ bidx,
                                                      const float* __restrict__ scsh,
                                                      float* __restrict__ out) {
    __shared__ float4 sc4[16], sh4[16];
    const int tid = threadIdx.x;
    const int bb  = blockIdx.x;
    const int bh  = bidx[bb * 3 + 1];
    const int bw  = bidx[bb * 3 + 2];
    if (tid < 16) {
        sc4[tid] = ((const float4*)scsh)[tid];
        sh4[tid] = ((const float4*)(scsh + CH))[tid];
    }
    __syncthreads();

    for (int e = tid; e < BOUTP * BOUTP * (CH / 4); e += 256) {
        const int c4 = e & 15;
        const int x  = (e >> 4) % BOUTP;
        const int y  = e / (BOUTP * 16);
        float4* p = (float4*)out +
                    ((size_t)(bh * BOUTP + y) * HW + (bw * BOUTP + x)) * (CH / 4) + c4;
        float4 q = *p;
        const float4 s = sc4[c4];
        const float4 t = sh4[c4];
        q.x = fmaxf(0.0f, q.x * s.x + t.x);
        q.y = fmaxf(0.0f, q.y * s.y + t.y);
        q.z = fmaxf(0.0f, q.z * s.z + t.z);
        q.w = fmaxf(0.0f, q.w * s.w + t.w);
        *p = q;
    }
}

// ---------------- launcher ----------------
extern "C" void kernel_launch(void* const* d_in, const int* in_sizes, int n_in,
                              void* d_out, int out_size, void* d_ws, size_t ws_size,
                              hipStream_t stream) {
    const float* inp    = (const float*)d_in[0];
    const int*   bidx   = (const int*)d_in[1];
    const float* conv_w = (const float*)d_in[2];
    const float* conv_b = (const float*)d_in[3];
    const float* gamma  = (const float*)d_in[4];
    const float* beta   = (const float*)d_in[5];
    float* out = (float*)d_out;

    float*          stats = (float*)d_ws;                              // 128 f32
    float*          scsh  = (float*)((char*)d_ws + 512);               // 128 f32
    unsigned short* Wt    = (unsigned short*)((char*)d_ws + 2048);     // 64x576 bf16

    hipMemsetAsync(d_out, 0, (size_t)out_size * sizeof(float), stream);
    hipMemsetAsync(d_ws, 0, 512, stream);

    pack_weights<<<(KTOT * CH + 255) / 256, 256, 0, stream>>>(conv_w, Wt);
    conv_wmma_kernel<<<NBLK, 256, 0, stream>>>(inp, bidx, Wt, conv_b, out, stats);
    bn_stats_kernel<<<1, 64, 0, stream>>>(stats, gamma, beta, scsh);
    bn_relu_kernel<<<NBLK, 256, 0, stream>>>(bidx, scsh, out);
}
